// MemoryLayer_90168543412401
// MI455X (gfx1250) — compile-verified, matrix operand order checked by reading, not run on hardware
//
#include <hip/hip_runtime.h>
#include <math.h>

typedef __attribute__((ext_vector_type(16))) __bf16 v16bf;
typedef __attribute__((ext_vector_type(8)))  __bf16 v8bf;
typedef __attribute__((ext_vector_type(8)))  float  v8f;

#define NB   2
#define SEQ  4096
#define DIN  512
#define WD   256
#define NH   4
#define DHD  64
#define MROWS (NB*SEQ)   // 8192

// K-index base for element-pair pp (0..7) of a 16-bit A/B WMMA fragment,
// per CDNA5 ISA 7.12.2 (lanes 0-15: half=0, lanes 16-31: half=1).
__device__ __host__ __forceinline__ int kbase_of(int pp, int half) {
    return (pp < 4) ? (pp * 2 + half * 8) : (16 + (pp - 4) * 2 + half * 8);
}

// Inverse map: k (0..31) -> (element e, half hh) within fragment.
__device__ __forceinline__ void kinv(int k, int& e, int& hh) {
    if (k < 16) { hh = k >> 3; int r = k & 7; e = 2 * (r >> 1) + (k & 1); }
    else { int k2 = k - 16; hh = k2 >> 3; int r = k2 & 7; e = 2 * (4 + (r >> 1)) + (k2 & 1); }
}

__device__ __forceinline__ v16bf frag_from(const __bf16* p, int half) {
    // A/B fragment from a row-major bf16 row: two contiguous 8-elem runs.
    v8bf lo = *(const v8bf*)(p + half * 8);
    v8bf hi = *(const v8bf*)(p + 16 + half * 8);
    return __builtin_shufflevector(lo, hi, 0,1,2,3,4,5,6,7,8,9,10,11,12,13,14,15);
}

__device__ __forceinline__ v16bf frag_from_swz(const __bf16* p) {
    // 16 contiguous bf16 (fragment-linear swizzled storage).
    v8bf lo = *(const v8bf*)(p);
    v8bf hi = *(const v8bf*)(p + 8);
    return __builtin_shufflevector(lo, hi, 0,1,2,3,4,5,6,7,8,9,10,11,12,13,14,15);
}

// CDNA5 async copy: one 16B global->LDS transfer, tracked with ASYNCcnt.
// GV addressing: 64-bit VGPR global address; VDST holds the LDS byte offset
// (low 32 bits of the LDS-aperture flat address, per ISA 10.2).
__device__ __forceinline__ void async_load_b128(const void* gptr, void* lptr) {
    unsigned lds = (unsigned)(size_t)lptr;
    asm volatile("global_load_async_to_lds_b128 %0, %1, off"
                 :: "v"(lds), "v"(gptr) : "memory");
}
__device__ __forceinline__ void wait_async0() {
    asm volatile("s_wait_asynccnt 0" ::: "memory");
}

// ---------------------------------------------------------------------------
// Pack f32 -> bf16 row-major (for x).
// ---------------------------------------------------------------------------
__global__ __launch_bounds__(256) void pack_bf16(const float* __restrict__ in,
                                                 __bf16* __restrict__ out, int n)
{
    int i = (blockIdx.x * 256 + threadIdx.x) * 8;
    if (i >= n) return;
    float4 a = *(const float4*)(in + i);
    float4 b = *(const float4*)(in + i + 4);
    v8bf o;
    o[0] = (__bf16)a.x; o[1] = (__bf16)a.y; o[2] = (__bf16)a.z; o[3] = (__bf16)a.w;
    o[4] = (__bf16)b.x; o[5] = (__bf16)b.y; o[6] = (__bf16)b.z; o[7] = (__bf16)b.w;
    *(v8bf*)(out + i) = o;
}

// ---------------------------------------------------------------------------
// Swizzle an f32 [K,N] weight into B-fragment-linear bf16.
// ---------------------------------------------------------------------------
__global__ __launch_bounds__(256) void swz_weight_bf16(const float* __restrict__ Wt,
                                                       __bf16* __restrict__ dst,
                                                       int K, int N)
{
    int t = blockIdx.x * 256 + threadIdx.x;
    int lanes = (K / 32) * (N / 16) * 32;
    if (t >= lanes) return;
    int lane = t & 31, tile = t >> 5;
    int tilesN = N >> 4;
    int nt = tile % tilesN, kt = tile / tilesN;
    int half = lane >> 4, ln = lane & 15;
    v16bf v;
#pragma unroll
    for (int pp = 0; pp < 8; ++pp) {
        int kb = kt * 32 + kbase_of(pp, half);
        int n  = nt * 16 + ln;
        v[2*pp+0] = (__bf16)Wt[(size_t)(kb + 0) * N + n];
        v[2*pp+1] = (__bf16)Wt[(size_t)(kb + 1) * N + n];
    }
    __bf16* o = dst + (size_t)t * 16;
    *(v8bf*)(o)     = __builtin_shufflevector(v, v, 0,1,2,3,4,5,6,7);
    *(v8bf*)(o + 8) = __builtin_shufflevector(v, v, 8,9,10,11,12,13,14,15);
}

// ---------------------------------------------------------------------------
// GEMM: C = act(A_bf16[M,K] @ Wswz[K,N] + bias).
// omode: 0 = bf16 row-major out, 1 = bf16 fragment-swizzled out, 2 = f32 (+resid)
// ---------------------------------------------------------------------------
__global__ __launch_bounds__(256) void gemm_bf16_wmma(
    const __bf16* __restrict__ Ab, const __bf16* __restrict__ Ws,
    const float* __restrict__ bias, const float* __restrict__ resid,
    void* __restrict__ Cout, int M, int N, int K, int act, int omode)
{
    const int lane = threadIdx.x & 31;
    const int wave = threadIdx.x >> 5;
    const int half = lane >> 4;
    const int ln   = lane & 15;

    const int ntn = N >> 6;
    const int wid = blockIdx.x * 8 + wave;
    const int tm  = wid / ntn;
    const int tn  = wid % ntn;
    if (tm * 16 >= M) return;

    const int row = tm * 16 + ln;
    const int tilesN = N >> 4;
    const __bf16* Arow = Ab + (size_t)row * K;
    v8f acc[4] = {};

    for (int k0 = 0; k0 < K; k0 += 32) {
        const v16bf a = frag_from(Arow + k0, half);
        if (k0 + 32 < K) __builtin_prefetch(Arow + k0 + 32, 0, 1);
        const int kt = k0 >> 5;
#pragma unroll
        for (int t = 0; t < 4; ++t) {
            const size_t tile = (size_t)kt * tilesN + (tn << 2) + t;
            const v16bf bm = frag_from_swz(Ws + (tile * 32 + lane) * 16);
            acc[t] = __builtin_amdgcn_wmma_f32_16x16x32_bf16(
                false, a, false, bm, (short)0, acc[t], false, false);
        }
    }

#pragma unroll
    for (int t = 0; t < 4; ++t) {
        const int n  = tn * 64 + t * 16 + ln;
        const float bv = bias ? bias[n] : 0.0f;
#pragma unroll
        for (int i = 0; i < 8; ++i) {
            const int r = tm * 16 + i + 8 * half;
            float v = acc[t][i] + bv;
            if (act == 1)
                v = 0.5f * v * (1.0f + erff(v * 0.70710678118654752f));
            if (omode == 2) {
                float* C = (float*)Cout;
                if (resid) v += resid[(size_t)r * N + n];
                C[(size_t)r * N + n] = v;
            } else if (omode == 0) {
                __bf16* C = (__bf16*)Cout;
                C[(size_t)r * N + n] = (__bf16)v;
            } else {
                __bf16* C = (__bf16*)Cout;
                const int mt = tm >> 1;
                const int k  = (tm & 1) * 16 + i + 8 * half;
                int e, hh; kinv(k, e, hh);
                const int lane2 = ln + 16 * hh;
                const size_t tile = (size_t)mt * tilesN + (tn << 2) + t;
                C[(tile * 32 + lane2) * 16 + e] = (__bf16)v;
            }
        }
    }
}

// ---------------------------------------------------------------------------
// Flash attention. Block = 8 waves sharing one (b,h) and a 128-row query
// stripe. K/V 32-key chunks are double-buffered in LDS via async global->LDS
// copies (once per block, not per wave). Each wave owns a 16-row query tile.
// ---------------------------------------------------------------------------
__global__ __launch_bounds__(256) void attn_flash_wmma(
    const __bf16* __restrict__ Qb, const __bf16* __restrict__ Kb,
    const __bf16* __restrict__ Vswz, __bf16* __restrict__ Ob)
{
    __shared__ __align__(16) __bf16 Ks[2][32][64];   // 8 KB, row-major keys
    __shared__ __align__(16) __bf16 Vs[2][2048];     // 8 KB, fragment-linear
    __shared__ __align__(16) __bf16 Pl[8][16][32];   // 8 KB, P relayout

    const int tid  = threadIdx.x;
    const int lane = tid & 31;
    const int wave = tid >> 5;
    const int half = lane >> 4;
    const int ln   = lane & 15;

    const int blk   = blockIdx.x;        // 256 blocks
    const int qs    = blk & 31;          // SEQ/128 query stripes
    const int bh    = blk >> 5;
    const int b     = bh >> 2;
    const int h     = bh & 3;
    const int qt    = qs * 8 + wave;     // this wave's 16-row tile
    const int tilesN = WD >> 4;

    const size_t rowbase = (size_t)b * SEQ;
    const int crow = tid >> 3, cseg = tid & 7;   // K copy: 32 rows x 8 x 16B

    auto issue_chunk = [&](int kc, int buf) {
        const __bf16* ksrc = Kb + (rowbase + kc + crow) * WD + h * DHD + cseg * 8;
        async_load_b128(ksrc, &Ks[buf][crow][cseg * 8]);
        const size_t mt = (rowbase + kc) >> 5;
        const __bf16* vsrc = Vswz + (mt * tilesN + h * 4) * 32 * 16 + tid * 8;
        async_load_b128(vsrc, &Vs[buf][tid * 8]);
    };

    // Q fragments (global, row-major bf16)
    const int qrow = qt * 16 + ln;
    const __bf16* qp = Qb + (rowbase + qrow) * WD + h * DHD;
    v16bf qa[2];
#pragma unroll
    for (int f = 0; f < 2; ++f) qa[f] = frag_from(qp + f * 32, half);

    float mrow[8], lrow[8];
    v8f acc[4] = {};
#pragma unroll
    for (int i = 0; i < 8; ++i) { mrow[i] = -INFINITY; lrow[i] = 0.0f; }
    const float scale = 0.125f;

    issue_chunk(0, 0);
    wait_async0();
    __syncthreads();

    for (int kc = 0, j = 0; kc < SEQ; kc += 32, ++j) {
        const int buf = j & 1;
        if (kc + 32 < SEQ) issue_chunk(kc + 32, buf ^ 1);

        // ---- scores: Q @ K_chunk^T (B-fragments from LDS) ----
        v8f s[2] = {};
#pragma unroll
        for (int sub = 0; sub < 2; ++sub) {
            const __bf16* kp = &Ks[buf][sub * 16 + ln][0];
#pragma unroll
            for (int f = 0; f < 2; ++f) {
                const v16bf bm = frag_from(kp + f * 32, half);
                s[sub] = __builtin_amdgcn_wmma_f32_16x16x32_bf16(
                    false, qa[f], false, bm, (short)0, s[sub], false, false);
            }
        }

        // ---- online softmax over the 32-key chunk ----
#pragma unroll
        for (int i = 0; i < 8; ++i) {
            float s0 = s[0][i] * scale;
            float s1 = s[1][i] * scale;
            float mx = fmaxf(s0, s1);
#pragma unroll
            for (int d = 1; d < 16; d <<= 1)
                mx = fmaxf(mx, __shfl_xor(mx, d, 16));
            const float mnew  = fmaxf(mrow[i], mx);
            const float alpha = __expf(mrow[i] - mnew);
            const float p0 = __expf(s0 - mnew);
            const float p1 = __expf(s1 - mnew);
            float rs = p0 + p1;
#pragma unroll
            for (int d = 1; d < 16; d <<= 1)
                rs += __shfl_xor(rs, d, 16);
            lrow[i] = lrow[i] * alpha + rs;
            mrow[i] = mnew;
#pragma unroll
            for (int t = 0; t < 4; ++t) acc[t][i] *= alpha;
            Pl[wave][i + 8 * half][ln]      = (__bf16)p0;
            Pl[wave][i + 8 * half][16 + ln] = (__bf16)p1;
        }
        asm volatile("s_wait_dscnt 0" ::: "memory");

        // ---- reload P as A-fragment (2x ds b128), then ctx += P @ V ----
        const v16bf pa = frag_from(&Pl[wave][ln][0], half);
#pragma unroll
        for (int t = 0; t < 4; ++t) {
            const v16bf vb = frag_from_swz(&Vs[buf][(t * 32 + lane) * 16]);
            acc[t] = __builtin_amdgcn_wmma_f32_16x16x32_bf16(
                false, pa, false, vb, (short)0, acc[t], false, false);
        }

        // next buffer resident + everyone done with current buffer
        wait_async0();
        __syncthreads();
    }

#pragma unroll
    for (int t = 0; t < 4; ++t) {
#pragma unroll
        for (int i = 0; i < 8; ++i) {
            const int r = qt * 16 + i + 8 * half;
            Ob[(rowbase + r) * WD + h * DHD + t * 16 + ln] =
                (__bf16)(acc[t][i] / lrow[i]);
        }
    }
}

// ---------------------------------------------------------------------------
extern "C" void kernel_launch(void* const* d_in, const int* in_sizes, int n_in,
                              void* d_out, int out_size, void* d_ws, size_t ws_size,
                              hipStream_t stream)
{
    const float* x = (const float*)d_in[0];
    const float* wsrc[11] = {
        (const float*)d_in[1],  (const float*)d_in[3],  (const float*)d_in[5],
        (const float*)d_in[7],  (const float*)d_in[9],  (const float*)d_in[11],
        (const float*)d_in[13], (const float*)d_in[15], (const float*)d_in[17],
        (const float*)d_in[19], (const float*)d_in[21] };
    const float* bsrc[11] = {
        (const float*)d_in[2],  (const float*)d_in[4],  (const float*)d_in[6],
        (const float*)d_in[8],  (const float*)d_in[10], (const float*)d_in[12],
        (const float*)d_in[14], (const float*)d_in[16], (const float*)d_in[18],
        (const float*)d_in[20], (const float*)d_in[22] };
    const int wK[11] = { DIN, DIN, DIN, WD, WD, WD, WD, WD, WD, WD, WD };
    const int wN[11] = { WD,  WD,  WD,  WD, WD, WD, WD, WD, WD, WD, DIN };
    float* out = (float*)d_out;

    // ---- workspace layout ----
    char* p = (char*)d_ws;
    const size_t ABYTES = (size_t)MROWS * WD * 2;
    __bf16* ab[4];
    for (int i = 0; i < 4; ++i) { ab[i] = (__bf16*)p; p += ABYTES; }
    __bf16* xb = (__bf16*)p; p += (size_t)MROWS * DIN * 2;
    __bf16* wswz[11];
    for (int i = 0; i < 11; ++i) {
        wswz[i] = (__bf16*)p;
        p += (size_t)wK[i] * wN[i] * 2;
    }

    // ---- stage 0: pack x, swizzle weights ----
    {
        const int n = MROWS * DIN;
        pack_bf16<<<n / (256 * 8), 256, 0, stream>>>(x, xb, n);
    }
    for (int i = 0; i < 11; ++i) {
        const int lanes = (wK[i] / 32) * (wN[i] / 16) * 32;
        swz_weight_bf16<<<(lanes + 255) / 256, 256, 0, stream>>>(
            wsrc[i], wswz[i], wK[i], wN[i]);
    }

    auto gemm = [&](const __bf16* A, int widx, const float* resid, void* C,
                    int act, int omode) {
        const int N = wN[widx], K = wK[widx];
        const int blocks = (MROWS / 16) * (N / 64) / 8;
        gemm_bf16_wmma<<<blocks, 256, 0, stream>>>(
            A, wswz[widx], bsrc[widx], resid, C, MROWS, N, K, act, omode);
    };

    // outer projections D -> W
    gemm(xb, 0, nullptr, ab[0], 0, 0);      // q0
    gemm(xb, 1, nullptr, ab[1], 0, 0);      // k0
    gemm(xb, 2, nullptr, ab[2], 0, 0);      // v0
    // MHA internal projections W -> W
    gemm(ab[0], 3, nullptr, ab[3], 0, 0);   // q   (row-major)
    gemm(ab[1], 4, nullptr, ab[0], 0, 0);   // k   (row-major)
    gemm(ab[2], 5, nullptr, ab[1], 0, 1);   // v   (fragment-swizzled)
    // flash attention -> ctx (bf16 row-major)
    attn_flash_wmma<<<256, 256, 0, stream>>>(ab[3], ab[0], ab[1], ab[2]);
    // attention output projection
    gemm(ab[2], 6, nullptr, ab[0], 0, 0);   // retrieved
    // deep MLP memory (exact gelu)
    gemm(ab[0], 7, nullptr, ab[1], 1, 0);   // h1
    gemm(ab[1], 8, nullptr, ab[2], 1, 0);   // h2
    gemm(ab[2], 9, nullptr, ab[0], 0, 0);   // mem
    // final projection W -> D with residual x (f32 out)
    gemm(ab[0], 10, x, out, 0, 2);
}
